// MultiHeadLatentAttention_27109833572426
// MI455X (gfx1250) — compile-verified, matrix operand order checked by reading, not run on hardware
//
#include <hip/hip_runtime.h>
#include <hip/hip_bf16.h>
#include <math.h>

#define BATCH 2
#define SEQ   2048
#define EMB   2048
#define NHEAD 16
#define ROWS  (BATCH*SEQ)   // 4096

typedef __attribute__((ext_vector_type(16))) _Float16 v16h;
typedef __attribute__((ext_vector_type(8)))  _Float16 v8h;
typedef __attribute__((ext_vector_type(8)))  float    v8f;

// ---------------------------------------------------------------------------
// Fragment load: 16-bit A/B matrix fragment for v_wmma_*_16x16x32_f16.
// Lane L holds row (L&15); K data in two contiguous 8-halfword chunks at
// k = g*8 and k = 16+g*8 (g = L>>4), per the CDNA5 16-bit A-matrix layout.
// Works for A (row-major [M,K]) and for B when stored transposed ([N,K]).
// ---------------------------------------------------------------------------
static __device__ inline v16h load_frag16(const _Float16* p, int ld) {
    const int lane = threadIdx.x & 31;
    const int r = lane & 15, g = lane >> 4;
    const _Float16* q = p + (size_t)r * ld + g * 8;
    v8h c0 = *(const v8h*)(q);
    v8h c1 = *(const v8h*)(q + 16);
    v16h a;
#pragma unroll
    for (int i = 0; i < 8; i++) { a[i] = c0[i]; a[i + 8] = c1[i]; }
    return a;
}

static __device__ inline v8f wmma_f16(v16h a, v16h b, v8f c) {
    // D = A(16x32) * B(32x16) + C ; f32 accumulate
    return __builtin_amdgcn_wmma_f32_16x16x32_f16(
        /*neg_a=*/false, a, /*neg_b=*/false, b,
        /*c_mod=*/(short)0, c, /*reuse_a=*/false, /*reuse_b=*/false);
}

// ---------------------------------------------------------------------------
// Generic WMMA GEMM: C[M,N] = A[M,K] * B[K,N], with B supplied TRANSPOSED
// (Bt[N,K], f16, row-major).  Block = 256 threads = 8 waves; each wave owns a
// 32x64 output tile (2 A-frags x 4 B-frags = 8 WMMAs per k-step), block
// covers 256x64.  The k-loop is unrolled by 2 with explicit ping-pong
// register buffers: loads for the next k-step are issued before the WMMA
// burst for the current one, and the two buffer sets alternate structurally
// so the compiler emits no copy-movs.
// blockIdx.z batches heads via the *Z strides.
// flags: bit0 = f16 output, bit1 = store C transposed (C[n][m]).
// Requires M%32==0, N%64==0, K%64==0 (true for every call here).
// ---------------------------------------------------------------------------
__global__ __launch_bounds__(256) void gemm_f16_wmma(
    const _Float16* __restrict__ A, const _Float16* __restrict__ Bt,
    void* __restrict__ Cv, int M, int N, int K, int lda, int ldb, int ldc,
    long long aZ, long long bZ, long long cZ, int flags)
{
    const int w  = threadIdx.x >> 5;
    const int m0 = blockIdx.y * 256 + w * 32;
    const int n0 = blockIdx.x * 64;
    const int z  = blockIdx.z;
    if (m0 >= M || n0 >= N) return;

    const _Float16* Ab = A  + (long long)z * aZ + (size_t)m0 * lda;
    const _Float16* Bb = Bt + (long long)z * bZ + (size_t)n0 * ldb;

    v8f acc[2][4] = {};
    v16h aP[2], bP[4], aQ[2], bQ[4];    // ping-pong fragment buffers

    auto load_set = [&](v16h (&aa)[2], v16h (&bb)[4], int kk) {
        aa[0] = load_frag16(Ab + kk, lda);
        aa[1] = load_frag16(Ab + (size_t)16 * lda + kk, lda);
#pragma unroll
        for (int j = 0; j < 4; j++)
            bb[j] = load_frag16(Bb + (size_t)(j * 16) * ldb + kk, ldb);
    };
    auto do_wmma = [&](v16h (&aa)[2], v16h (&bb)[4]) {
#pragma unroll
        for (int j = 0; j < 4; j++) {
            acc[0][j] = wmma_f16(aa[0], bb[j], acc[0][j]);
            acc[1][j] = wmma_f16(aa[1], bb[j], acc[1][j]);
        }
    };

    load_set(aP, bP, 0);
    for (int k = 0; k < K; k += 64) {
        if (K >= 512) {   // long-K GEMMs: prefetch ahead into L2/L0
            __builtin_prefetch(Ab + k + 128, 0, 1);
            __builtin_prefetch(Bb + k + 128, 0, 1);
        }
        load_set(aQ, bQ, k + 32);                        // k+32 <= K-32 always
        do_wmma(aP, bP);
        load_set(aP, bP, (k + 64 < K) ? k + 64 : 0);     // clamped safe index
        do_wmma(aQ, bQ);
    }

    const int lane = threadIdx.x & 31;
    const int n = lane & 15, g = lane >> 4;
#pragma unroll
    for (int i = 0; i < 2; i++) {
#pragma unroll
        for (int j = 0; j < 4; j++) {
            const int col = n0 + j * 16 + n;
#pragma unroll
            for (int r = 0; r < 8; r++) {
                const int row = m0 + i * 16 + r + g * 8;
                const float val = acc[i][j][r];
                if (flags & 2) {
                    ((_Float16*)Cv)[(long long)z * cZ + (size_t)col * ldc + row] = (_Float16)val;
                } else if (flags & 1) {
                    ((_Float16*)Cv)[(long long)z * cZ + (size_t)row * ldc + col] = (_Float16)val;
                } else {
                    ((float*)Cv)[(long long)z * cZ + (size_t)row * ldc + col] = val;
                }
            }
        }
    }
}

// ---------------------------------------------------------------------------
// Conversion kernels
// ---------------------------------------------------------------------------
__global__ __launch_bounds__(256) void cvt_f32_f16(const float* __restrict__ src,
                                                   _Float16* __restrict__ dst, int nElem)
{
    int t = blockIdx.x * 256 + threadIdx.x;
    if (t < nElem) dst[t] = (_Float16)src[t];
}

// dst[z][rowOff+n][k] (f16, ld=dstLd) = src[z][k][n] (f32)
__global__ __launch_bounds__(256) void cvt_transpose(
    const float* __restrict__ src, _Float16* __restrict__ dst,
    int K, int N, long long srcZ, long long dstZ, int dstLd, int rowOff)
{
    int t = blockIdx.x * 256 + threadIdx.x;
    if (t >= K * N) return;
    int k = t / N, n = t % N;
    int z = blockIdx.z;
    dst[(long long)z * dstZ + (size_t)(rowOff + n) * dstLd + k] =
        (_Float16)src[(long long)z * srcZ + (size_t)k * N + n];
}

// ---------------------------------------------------------------------------
// RoPE (interleaved pairs, half = 32, base 10000) applied to f16 buffers.
// ---------------------------------------------------------------------------
__global__ __launch_bounds__(256) void rope_kr_kernel(const _Float16* __restrict__ src,
                                                      _Float16* __restrict__ dst)
{
    int t = blockIdx.x * 256 + threadIdx.x;
    if (t >= ROWS * 32) return;
    int row = t >> 5, i = t & 31;
    int s = row & (SEQ - 1);
    float freq = __expf((float)i * (-9.210340371976184f / 32.f)); // -ln(1e4)/32
    float ang = (float)s * freq;
    float c = __cosf(ang), sn = __sinf(ang);
    const _Float16* p = src + (size_t)row * 320 + 256 + 2 * i;
    float x0 = (float)p[0], x1 = (float)p[1];
    dst[(size_t)row * 64 + 2 * i]     = (_Float16)(x0 * c - x1 * sn);
    dst[(size_t)row * 64 + 2 * i + 1] = (_Float16)(x0 * sn + x1 * c);
}

__global__ __launch_bounds__(256) void rope_q_kernel(_Float16* __restrict__ q)
{
    int t = blockIdx.x * 256 + threadIdx.x;
    if (t >= NHEAD * ROWS * 32) return;
    int i = t & 31;
    int row = (t >> 5) & (ROWS - 1);
    int h = t >> 17;                 // 32 * 4096 = 2^17
    int s = row & (SEQ - 1);
    float freq = __expf((float)i * (-9.210340371976184f / 32.f));
    float ang = (float)s * freq;
    float c = __cosf(ang), sn = __sinf(ang);
    _Float16* p = q + ((size_t)h * ROWS + row) * 128 + 64 + 2 * i;
    float x0 = (float)p[0], x1 = (float)p[1];
    p[0] = (_Float16)(x0 * c - x1 * sn);
    p[1] = (_Float16)(x0 * sn + x1 * c);
}

// ---------------------------------------------------------------------------
// Flash attention.  One 16-row query tile per wave, 8 waves / block.
// grid = (SEQ/128, BATCH, NHEAD).
//   qbuf : [H][ROWS][128]  cols 0..63 = q_nope, 64..127 = q_rope (f16)
//   kn   : [H][ROWS][64]   k_nope (f16)
//   kr   : [ROWS][64]      k_rope (f16, shared across heads)
//   vt   : [H][128][ROWS]  V transposed (f16)
//   aout : [ROWS][EMB]     feature = h*128 + d  (f16)
// V fragments for the chunk are loaded BEFORE the softmax so their latency
// hides behind the exp/shuffle VALU work.
// ---------------------------------------------------------------------------
__global__ __launch_bounds__(256) void mla_flash(
    const _Float16* __restrict__ qbuf, const _Float16* __restrict__ kn,
    const _Float16* __restrict__ kr,   const _Float16* __restrict__ vt,
    _Float16* __restrict__ aout)
{
    __shared__ alignas(16) _Float16 P[8][16][32];   // per-wave P tile (f16)

    const int w    = threadIdx.x >> 5;
    const int lane = threadIdx.x & 31;
    const int n = lane & 15, g = lane >> 4;
    const int b = blockIdx.y, h = blockIdx.z;
    const int q0 = blockIdx.x * 128 + w * 16;
    const float scale = 0.08838834764831845f;       // 1/sqrt(d_h=128)

    const _Float16* Qb = qbuf + (size_t)h * ROWS * 128 + (size_t)(b * SEQ + q0) * 128;
    const v16h qn0 = load_frag16(Qb + 0,  128);
    const v16h qn1 = load_frag16(Qb + 32, 128);
    const v16h qr0 = load_frag16(Qb + 64, 128);
    const v16h qr1 = load_frag16(Qb + 96, 128);

    const _Float16* Kb = kn + (size_t)h * ROWS * 64 + (size_t)(b * SEQ) * 64;
    const _Float16* Rb = kr + (size_t)(b * SEQ) * 64;
    const _Float16* Vb = vt + (size_t)h * 128 * ROWS + (size_t)(b * SEQ);

    v8f o[8] = {};
    float mrow[8], lrow[8];
#pragma unroll
    for (int r = 0; r < 8; r++) { mrow[r] = -1e30f; lrow[r] = 0.f; }

    for (int tb = 0; tb <= q0 + 15; tb += 32) {
        // ---- scores for two 16-key tiles (d_c=64 nope + d_r=64 rope) ----
        v8f s0 = {}, s1 = {};
        {
            v16h k0a = load_frag16(Kb + (size_t)tb * 64 + 0,  64);
            v16h k0b = load_frag16(Kb + (size_t)tb * 64 + 32, 64);
            v16h r0a = load_frag16(Rb + (size_t)tb * 64 + 0,  64);
            v16h r0b = load_frag16(Rb + (size_t)tb * 64 + 32, 64);
            v16h k1a = load_frag16(Kb + (size_t)(tb + 16) * 64 + 0,  64);
            v16h k1b = load_frag16(Kb + (size_t)(tb + 16) * 64 + 32, 64);
            v16h r1a = load_frag16(Rb + (size_t)(tb + 16) * 64 + 0,  64);
            v16h r1b = load_frag16(Rb + (size_t)(tb + 16) * 64 + 32, 64);
            s0 = wmma_f16(qn0, k0a, s0); s0 = wmma_f16(qn1, k0b, s0);
            s0 = wmma_f16(qr0, r0a, s0); s0 = wmma_f16(qr1, r0b, s0);
            s1 = wmma_f16(qn0, k1a, s1); s1 = wmma_f16(qn1, k1b, s1);
            s1 = wmma_f16(qr0, r1a, s1); s1 = wmma_f16(qr1, r1b, s1);
        }

        // ---- V fragments early: latency hidden behind softmax VALU ----
        v16h bv[8];
#pragma unroll
        for (int j = 0; j < 8; j++)
            bv[j] = load_frag16(Vb + (size_t)(j * 16) * ROWS + tb, ROWS);

        // ---- causal mask + online softmax (row reductions over 16 lanes) ----
#pragma unroll
        for (int r = 0; r < 8; r++) {
            const int row  = q0 + r + g * 8;
            const int key0 = tb + n, key1 = key0 + 16;
            float e0 = (key0 <= row) ? s0[r] * scale : -1e30f;
            float e1 = (key1 <= row) ? s1[r] * scale : -1e30f;
            float mx = fmaxf(e0, e1);
#pragma unroll
            for (int off = 1; off < 16; off <<= 1) mx = fmaxf(mx, __shfl_xor(mx, off, 16));
            const float mnew  = fmaxf(mrow[r], mx);
            const float alpha = __expf(mrow[r] - mnew);
            const float p0 = __expf(e0 - mnew);
            const float p1 = __expf(e1 - mnew);
            float ps = p0 + p1;
#pragma unroll
            for (int off = 1; off < 16; off <<= 1) ps += __shfl_xor(ps, off, 16);
            lrow[r] = lrow[r] * alpha + ps;
            mrow[r] = mnew;
#pragma unroll
            for (int j = 0; j < 8; j++) o[j][r] *= alpha;
            // Stage P (D-fragment layout) into LDS for A-fragment reload.
            P[w][r + g * 8][n]      = (_Float16)p0;
            P[w][r + g * 8][16 + n] = (_Float16)p1;
        }

        // ---- O += P(16x32) * V(32x128); same-wave LDS ops are in-order ----
        const v16h pa = load_frag16(&P[w][0][0], 32);
#pragma unroll
        for (int j = 0; j < 8; j++)
            o[j] = wmma_f16(pa, bv[j], o[j]);
    }

    // ---- normalize and store head-concatenated f16 output ----
#pragma unroll
    for (int r = 0; r < 8; r++) {
        const float inv = 1.f / lrow[r];
        const size_t row = (size_t)(b * SEQ + q0 + r + g * 8);
#pragma unroll
        for (int j = 0; j < 8; j++)
            aout[row * EMB + h * 128 + j * 16 + n] = (_Float16)(o[j][r] * inv);
    }
}

// ---------------------------------------------------------------------------
// Host orchestration
// ---------------------------------------------------------------------------
extern "C" void kernel_launch(void* const* d_in, const int* in_sizes, int n_in,
                              void* d_out, int out_size, void* d_ws, size_t ws_size,
                              hipStream_t stream)
{
    const float* x    = (const float*)d_in[0];
    const float* Wdkv = (const float*)d_in[1];
    const float* Wdq  = (const float*)d_in[2];
    const float* Wkr  = (const float*)d_in[3];
    const float* Wqc  = (const float*)d_in[4];
    const float* Wqr  = (const float*)d_in[5];
    const float* Wuk  = (const float*)d_in[6];
    const float* Wuv  = (const float*)d_in[7];
    const float* Wo   = (const float*)d_in[8];
    float* out = (float*)d_out;

    char* ws = (char*)d_ws;
    size_t off = 0;
    auto alloc = [&](size_t elems) {
        _Float16* p = (_Float16*)(ws + off);
        off += ((elems * sizeof(_Float16) + 255) & ~(size_t)255);
        return p;
    };
    _Float16* xh    = alloc((size_t)ROWS * EMB);        // x in f16 (reused as aout)
    _Float16* wcat  = alloc((size_t)320 * EMB);         // [dkv|dq|kr]^T
    _Float16* ckv   = alloc((size_t)ROWS * 320);        // c_kv | x_dq | kr_raw
    _Float16* krb   = alloc((size_t)ROWS * 64);         // roped k_rope
    _Float16* wqt   = alloc((size_t)NHEAD * 128 * 128); // [Wqc|Wqr]^T per head
    _Float16* wukt  = alloc((size_t)NHEAD * 64 * 128);
    _Float16* wuvt  = alloc((size_t)NHEAD * 128 * 128);
    _Float16* qbuf  = alloc((size_t)NHEAD * ROWS * 128);
    _Float16* knope = alloc((size_t)NHEAD * ROWS * 64);
    _Float16* vt    = alloc((size_t)NHEAD * 128 * ROWS);
    _Float16* wot   = alloc((size_t)EMB * EMB);
    _Float16* aout  = xh;   // xh is dead after GEMM1; reuse for attention output

    const dim3 blk(256);

    // ---- precision conversion / weight transposition ----
    cvt_f32_f16<<<(ROWS * EMB + 255) / 256, blk, 0, stream>>>(x, xh, ROWS * EMB);
    cvt_transpose<<<dim3((EMB * 128 + 255) / 256, 1, 1), blk, 0, stream>>>(Wdkv, wcat, EMB, 128, 0, 0, EMB, 0);
    cvt_transpose<<<dim3((EMB * 128 + 255) / 256, 1, 1), blk, 0, stream>>>(Wdq,  wcat, EMB, 128, 0, 0, EMB, 128);
    cvt_transpose<<<dim3((EMB * 64  + 255) / 256, 1, 1), blk, 0, stream>>>(Wkr,  wcat, EMB, 64,  0, 0, EMB, 256);
    cvt_transpose<<<dim3((128 * 64  + 255) / 256, 1, NHEAD), blk, 0, stream>>>(Wqc, wqt,  128, 64,  128 * 64,  128 * 128, 128, 0);
    cvt_transpose<<<dim3((128 * 64  + 255) / 256, 1, NHEAD), blk, 0, stream>>>(Wqr, wqt,  128, 64,  128 * 64,  128 * 128, 128, 64);
    cvt_transpose<<<dim3((128 * 64  + 255) / 256, 1, NHEAD), blk, 0, stream>>>(Wuk, wukt, 128, 64,  128 * 64,  64 * 128,  128, 0);
    cvt_transpose<<<dim3((128 * 128 + 255) / 256, 1, NHEAD), blk, 0, stream>>>(Wuv, wuvt, 128, 128, 128 * 128, 128 * 128, 128, 0);
    cvt_transpose<<<dim3((EMB * EMB + 255) / 256, 1, 1), blk, 0, stream>>>(Wo, wot, EMB, EMB, 0, 0, EMB, 0);

    // ---- GEMM1: [4096x2048] x [2048x320] -> c_kv | x_dq | k_rope_raw ----
    gemm_f16_wmma<<<dim3(320 / 64, ROWS / 256, 1), blk, 0, stream>>>(
        xh, wcat, ckv, ROWS, 320, EMB, EMB, EMB, 320, 0, 0, 0, 1);
    rope_kr_kernel<<<(ROWS * 32 + 255) / 256, blk, 0, stream>>>(ckv, krb);

    // ---- per-head projections (blockIdx.z = head) ----
    gemm_f16_wmma<<<dim3(128 / 64, ROWS / 256, NHEAD), blk, 0, stream>>>(
        ckv + 128, wqt, qbuf, ROWS, 128, 128, 320, 128, 128,
        0, 128 * 128, (long long)ROWS * 128, 1);
    rope_q_kernel<<<(NHEAD * ROWS * 32 + 255) / 256, blk, 0, stream>>>(qbuf);
    gemm_f16_wmma<<<dim3(64 / 64, ROWS / 256, NHEAD), blk, 0, stream>>>(
        ckv, wukt, knope, ROWS, 64, 128, 320, 128, 64,
        0, 64 * 128, (long long)ROWS * 64, 1);
    gemm_f16_wmma<<<dim3(128 / 64, ROWS / 256, NHEAD), blk, 0, stream>>>(
        ckv, wuvt, vt, ROWS, 128, 128, 320, 128, ROWS,
        0, 128 * 128, (long long)128 * ROWS, 1 | 2);   // V stored transposed

    // ---- flash attention ----
    mla_flash<<<dim3(SEQ / 128, BATCH, NHEAD), blk, 0, stream>>>(qbuf, knope, krb, vt, aout);

    // ---- output projection: [4096x2048] x Wo -> f32 out ----
    gemm_f16_wmma<<<dim3(EMB / 64, ROWS / 256, 1), blk, 0, stream>>>(
        aout, wot, out, ROWS, EMB, EMB, EMB, EMB, EMB, 0, 0, 0, 0);
}
